// BioEncoder_68135361183825
// MI455X (gfx1250) — matrix operand order repaired and checked by reading, not compiled, not run on hardware
//
#include <hip/hip_runtime.h>
#include <cfloat>

// ---------------------------------------------------------------------------
// BioEncoder (GIN x3 + pooling + mic/dis branches) for MI455X (gfx1250).
//
// GEMMs: v_wmma_f32_16x16x32_f16 with f16 weights pre-transposed so B
// fragments are two ds_load_b128.  W tiles are staged into LDS by the Tensor
// Data Mover (tensor_load_to_lds + s_wait_tensorcnt, 6-arg clang-23 form),
// with guarded async/sync fallbacks.
// Memory-bound parts (edge scatter-add, batchnorm stats, segment-max pool)
// use coalesced float4 traffic and f32/u32 global atomics.
//
// Assumed d_in layout (top-level dict insertion order; params flattened as a
// JAX pytree, i.e. recursively sorted keys; 'W' sorts before 'b'):
//   0:x_d  1:mic_feature  2:dis_feature
//   3:bn1.b 4:bn1.g 5:bn2.b 6:bn2.g 7:bn3.b 8:bn3.g
//   9:bn_dis.b 10:bn_dis.g 11:bn_mic.b 12:bn_mic.g
//   13:conv1.Wa 14:conv1.Wb 15:conv1.ba 16:conv1.bb
//   17:conv2.Wa 18:conv2.Wb 19:conv2.ba 20:conv2.bb
//   21:conv3.Wa 22:conv3.Wb 23:conv3.ba 24:conv3.bb
//   25:dis.W 26:dis.b 27:fc1_xd.W 28:fc1_xd.b 29:mic.W 30:mic.b
//   31:edge_index (2 x 1.6M int32)  32:batch (400k int32)
// d_out = [x (12800x128) | x_mic (4096x128) | x_dis (4096x128)] fp32.
// ---------------------------------------------------------------------------

#define N_NODES   400000
#define N_EDGES   1600000
#define N_GRAPHS  12800
#define ODIM      128
#define BN_EPS    1e-5f
#define LDS_KPAD  40            // halves per LDS row (80B; 20*n mod 64 distinct)

typedef __attribute__((ext_vector_type(16))) _Float16 v16h;
typedef __attribute__((ext_vector_type(8)))  _Float16 v8h;
typedef __attribute__((ext_vector_type(8)))  float    v8f;

#ifndef __has_builtin
#define __has_builtin(x) 0
#endif

#if __has_builtin(__builtin_amdgcn_tensor_load_to_lds) && __has_builtin(__builtin_amdgcn_s_wait_tensorcnt)
#  define W_TILE_TDM 1
#else
#  define W_TILE_TDM 0
#endif
#if !W_TILE_TDM && __has_builtin(__builtin_amdgcn_global_load_async_to_lds_b128)
#  define W_TILE_ASYNC 1
#else
#  define W_TILE_ASYNC 0
#endif
#if !W_TILE_TDM && !W_TILE_ASYNC && __has_builtin(__builtin_amdgcn_load_to_lds)
#  define W_TILE_L2L 1
#else
#  define W_TILE_L2L 0
#endif
#if !W_TILE_TDM && !W_TILE_ASYNC && !W_TILE_L2L && __has_builtin(__builtin_amdgcn_global_load_lds)
#  define W_TILE_GLL 1
#else
#  define W_TILE_GLL 0
#endif

#if W_TILE_TDM
typedef unsigned int u32x4 __attribute__((ext_vector_type(4)));
typedef int          i32x4 __attribute__((ext_vector_type(4)));
typedef int          i32x8 __attribute__((ext_vector_type(8)));
#endif

// ------------------------- weight pre-conversion ---------------------------
// WhT[n*K + k] = f16( W[k*128 + n] ) for k < Ksrc, else 0.  (K mult of 32)
__global__ void convert_wT_kernel(const float* __restrict__ W,
                                  _Float16* __restrict__ WhT, int K, int Ksrc)
{
    const int i = blockIdx.x * blockDim.x + threadIdx.x;     // over 128*K
    if (i < ODIM * K) {
        const int n = i / K, k = i - n * K;
        WhT[i] = (k < Ksrc) ? (_Float16)W[(size_t)k * ODIM + n] : (_Float16)0.f;
    }
}

// -------------------------------- GEMM -------------------------------------
// OUT[M x 128] = act( A[M x K] @ W[K x 128] + bias ), act = relu if do_relu.
// WhT is the f16 transposed weight: WhT[n][k], row stride K halves.
// Block = 256 threads = 8 wave32 waves; each wave computes a 16-row x 128-col
// strip; block covers 128 rows.  K is a multiple of 32.
__global__ __launch_bounds__(256)
void gemm_wmma_kernel(const float* __restrict__ A, int lda,
                      const _Float16* __restrict__ WhT,   // 128 x K (transposed)
                      const float* __restrict__ bias,     // 128
                      float* __restrict__ OUT, int ldo,
                      int M, int K, int do_relu)
{
    __shared__ _Float16 Wlds[ODIM][LDS_KPAD];   // tile: Wlds[n][kk], kk=0..31

    const int lane = threadIdx.x & 31;
    const int wave = threadIdx.x >> 5;
    const int rowBase = blockIdx.x * 128 + wave * 16;

    // A fragment row for this lane (A lanes hold row M = lane%16).
    int arowIdx = rowBase + (lane & 15);
    if (arowIdx >= M) arowIdx = M - 1;               // clamp (M mult of 128 here)
    const float* __restrict__ arow = A + (size_t)arowIdx * lda;

    // 16-bit A(16x32) lane layout: lanes 0-15 hold K {0..7,16..23},
    // lanes 16-31 hold K {8..15,24..31}.
    const int kgrpA = (lane >> 4) * 8;
    // 16-bit B(32x16) lane layout: lanes 0-15 hold K {0..7,16..23},
    // lanes 16-31 hold K {8..15,24..31}; column n = lane%16.
    const int kgrpB = (lane >> 4) * 8;
    const int ncol  = lane & 15;

    v8f acc[8] = {};   // 8 N-tiles of 16 cols each => full 128-wide strip

    for (int kt = 0; kt < K; kt += 32) {
        // ---- stage the 128x32-half W tile into LDS ----
#if W_TILE_TDM
        // Tensor Data Mover: one descriptor per tile, issued by wave 0.
        // D# per CDNA5 ISA: 2D tile 32x128 of 2-byte
        // elements, row stride = K elements, LDS padding 4 DWORDs (8 halves)
        // after every 16 DWORDs (32 halves) -> 40-half LDS row stride.
        if (threadIdx.x < 32) {
            const unsigned long long gaddr =
                (unsigned long long)(WhT + (size_t)kt);          // tile start (bytes)
            const unsigned ldsaddr =
                (unsigned)(unsigned long long)(void*)&Wlds[0][0]; // flat->LDS offset
            const u32x4 g0 = {
                1u,                                              // count=1, user D#
                ldsaddr,                                         // lds_addr
                (unsigned)(gaddr & 0xFFFFFFFFull),               // global_addr[31:0]
                (unsigned)((gaddr >> 32) & 0x01FFFFFFull)        // global_addr[56:32]
                    | 0x80000000u                                // type=2 ("image")
            };
            const i32x8 g1 = {
                (int)((1u << 16)            // data_size = 2 bytes
                    | (1u << 20)            // pad_enable
                    | (3u << 22)            // pad_interval: 16 DWORDs (64B)
                    | (3u << 25)),          // pad_amount:   4 DWORDs (16B)
                (int)(32u << 16),           // tensor_dim0 = 32   (bits 79:48)
                (int)(128u << 16),          // tensor_dim1 = 128  (bits 111:80)
                (int)(32u << 16),           // tile_dim0 = 32     (bits 127:112)
                (int)128,                   // tile_dim1 = 128    (bits 143:128)
                (int)K,                     // tensor_dim0_stride = K elements
                0, 0                        // dim1 stride unused (2D)
            };
            const i32x4 gz4 = {0, 0, 0, 0};
            const i32x8 gz8 = {0, 0, 0, 0, 0, 0, 0, 0};
            __builtin_amdgcn_tensor_load_to_lds(g0, g1, gz4, gz4, gz8, 0);
        }
        __builtin_amdgcn_s_wait_tensorcnt(0);
        __syncthreads();
#else
        for (int i = threadIdx.x; i < 512; i += 256) {       // uniform: 2 iters
            const int n = i >> 2, j = i & 3;                 // j: 8-half chunk
            const _Float16* gsrc = WhT + (size_t)n * K + kt + j * 8;
            _Float16*       ldst = &Wlds[n][j * 8];
#  if W_TILE_ASYNC
            __builtin_amdgcn_global_load_async_to_lds_b128(
                (const __attribute__((address_space(1))) void*)gsrc,
                (__attribute__((address_space(3))) void*)ldst, 0, 0);
#  elif W_TILE_L2L
            __builtin_amdgcn_load_to_lds(
                (void*)gsrc, (__attribute__((address_space(3))) void*)ldst, 16, 0, 0);
#  elif W_TILE_GLL
            __builtin_amdgcn_global_load_lds(
                (const __attribute__((address_space(1))) void*)gsrc,
                (__attribute__((address_space(3))) void*)ldst, 16, 0, 0);
#  else
            *(v8h*)ldst = *(const v8h*)gsrc;
#  endif
        }
#  if W_TILE_ASYNC || W_TILE_L2L || W_TILE_GLL
#    if __has_builtin(__builtin_amdgcn_s_wait_asynccnt)
        __builtin_amdgcn_s_wait_asynccnt(0);
#    else
        asm volatile("s_wait_asynccnt 0" ::: "memory");
#    endif
#  endif
        __syncthreads();
#endif

        // ---- A fragment: 4x global_load_b128 + f32->f16 convert ----
        const float4 a0 = *(const float4*)(arow + kt + kgrpA);
        const float4 a1 = *(const float4*)(arow + kt + kgrpA + 4);
        const float4 a2 = *(const float4*)(arow + kt + 16 + kgrpA);
        const float4 a3 = *(const float4*)(arow + kt + 16 + kgrpA + 4);
        v16h afrag;
        afrag[0]  = (_Float16)a0.x; afrag[1]  = (_Float16)a0.y;
        afrag[2]  = (_Float16)a0.z; afrag[3]  = (_Float16)a0.w;
        afrag[4]  = (_Float16)a1.x; afrag[5]  = (_Float16)a1.y;
        afrag[6]  = (_Float16)a1.z; afrag[7]  = (_Float16)a1.w;
        afrag[8]  = (_Float16)a2.x; afrag[9]  = (_Float16)a2.y;
        afrag[10] = (_Float16)a2.z; afrag[11] = (_Float16)a2.w;
        afrag[12] = (_Float16)a3.x; afrag[13] = (_Float16)a3.y;
        afrag[14] = (_Float16)a3.z; afrag[15] = (_Float16)a3.w;

#pragma unroll
        for (int nt = 0; nt < 8; ++nt) {
            const int n = nt * 16 + ncol;
            const v8h b0 = *(const v8h*)(&Wlds[n][kgrpB]);      // 2x ds_load_b128
            const v8h b1 = *(const v8h*)(&Wlds[n][kgrpB + 16]);
            const v16h bfrag = __builtin_shufflevector(
                b0, b1, 0, 1, 2, 3, 4, 5, 6, 7, 8, 9, 10, 11, 12, 13, 14, 15);
            acc[nt] = __builtin_amdgcn_wmma_f32_16x16x32_f16(
                false, afrag, false, bfrag, (short)0, acc[nt], false, false);
        }
        __syncthreads();
    }

    // Epilogue: C/D layout — lane holds col n = lane%16; VGPR r holds row
    // M = rowBase + r + 8*(lane/16).
    const int rOff = (lane >> 4) * 8;
#pragma unroll
    for (int nt = 0; nt < 8; ++nt) {
        const int col = nt * 16 + ncol;
        const float bv = bias[col];
#pragma unroll
        for (int r = 0; r < 8; ++r) {
            const int row = rowBase + rOff + r;
            if (row < M) {
                float v = acc[nt][r] + bv;
                if (do_relu) v = fmaxf(v, 0.f);
                OUT[(size_t)row * ldo + col] = v;
            }
        }
    }
}

// --------------------------- conv1 helpers ---------------------------------
__global__ void init_agg78_kernel(const float* __restrict__ xd, float* __restrict__ agg)
{
    const long long i = (long long)blockIdx.x * blockDim.x + threadIdx.x;
    if (i < (long long)N_NODES * ODIM) {
        const int c = (int)(i & 127);
        const long long n = i >> 7;
        agg[i] = (c < 78) ? xd[n * 78 + c] : 0.f;
    }
}

__global__ void scatter78_kernel(const int* __restrict__ src, const int* __restrict__ dst,
                                 const float* __restrict__ xd, float* __restrict__ agg)
{
    const long long i = (long long)blockIdx.x * blockDim.x + threadIdx.x;
    const long long e = i >> 7;
    const int c = (int)(i & 127);
    if (e < N_EDGES && c < 78)
        atomicAdd(&agg[(long long)dst[e] * ODIM + c], xd[(long long)src[e] * 78 + c]);
}

__global__ void scatter128_kernel(const int* __restrict__ src, const int* __restrict__ dst,
                                  const float* __restrict__ x, float* __restrict__ agg)
{
    const long long i = (long long)blockIdx.x * blockDim.x + threadIdx.x;
    const long long e = i >> 5;
    if (e < N_EDGES) {
        const int c4 = (int)(i & 31) * 4;
        const float4 v = *(const float4*)(x + (long long)src[e] * ODIM + c4);
        float* p = agg + (long long)dst[e] * ODIM + c4;
        atomicAdd(p + 0, v.x); atomicAdd(p + 1, v.y);
        atomicAdd(p + 2, v.z); atomicAdd(p + 3, v.w);
    }
}

// ------------------------------ batchnorm ----------------------------------
__global__ void zero_stats_kernel(float* __restrict__ stats)
{
    if (threadIdx.x < 256) stats[threadIdx.x] = 0.f;
}

// Column sums of relu(pre) and relu(pre)^2; 128 threads/block, one column per
// thread; rows strided by block; block partials merged with atomics.
__global__ void bn_stats_relu_kernel(const float* __restrict__ pre,
                                     float* __restrict__ stats, int M, int rowsPerBlock)
{
    const int c  = threadIdx.x;                 // 0..127
    const int r0 = blockIdx.x * rowsPerBlock;
    int rend = r0 + rowsPerBlock; if (rend > M) rend = M;
    float s = 0.f, s2 = 0.f;
    for (int r = r0; r < rend; ++r) {
        const float v = fmaxf(pre[(size_t)r * ODIM + c], 0.f);
        s += v; s2 += v * v;
    }
    atomicAdd(&stats[c],       s);
    atomicAdd(&stats[128 + c], s2);
}

__global__ void bn_apply_relu_kernel(const float* __restrict__ pre,
                                     const float* __restrict__ stats,
                                     const float* __restrict__ g,
                                     const float* __restrict__ b,
                                     float* __restrict__ out, int M)
{
    const long long i = (long long)blockIdx.x * blockDim.x + threadIdx.x;
    if (i < (long long)M * ODIM) {
        const int c = (int)(i & 127);
        const float invM = 1.f / (float)M;
        const float mean = stats[c] * invM;
        const float var  = stats[128 + c] * invM - mean * mean;
        const float v    = fmaxf(pre[i], 0.f);
        out[i] = g[c] * (v - mean) * rsqrtf(var + BN_EPS) + b[c];
    }
}

// ---------------------------- segment max pool -----------------------------
__device__ __forceinline__ unsigned enc_f32(float f)
{
    const unsigned u = __float_as_uint(f);
    return (u & 0x80000000u) ? ~u : (u | 0x80000000u);    // order-preserving
}
__device__ __forceinline__ float dec_f32(unsigned u)
{
    return (u & 0x80000000u) ? __uint_as_float(u & 0x7FFFFFFFu)
                             : __uint_as_float(~u);
}

__global__ void pool_init_kernel(unsigned* __restrict__ pooled)
{
    const int i = blockIdx.x * blockDim.x + threadIdx.x;
    if (i < N_GRAPHS * ODIM) pooled[i] = enc_f32(-FLT_MAX);
}

__global__ void pool_max_kernel(const float* __restrict__ x,
                                const int* __restrict__ batch,
                                unsigned* __restrict__ pooled)
{
    const long long i = (long long)blockIdx.x * blockDim.x + threadIdx.x;
    if (i < (long long)N_NODES * ODIM) {
        const int c = (int)(i & 127);
        const long long n = i >> 7;
        atomicMax(&pooled[(long long)batch[n] * ODIM + c], enc_f32(x[i]));
    }
}

__global__ void pool_decode_kernel(unsigned* __restrict__ pooled)
{
    const int i = blockIdx.x * blockDim.x + threadIdx.x;
    if (i < N_GRAPHS * ODIM) {
        const float f = dec_f32(pooled[i]);
        ((float*)pooled)[i] = f;
    }
}

// ------------------------------- driver ------------------------------------
extern "C" void kernel_launch(void* const* d_in, const int* in_sizes, int n_in,
                              void* d_out, int out_size, void* d_ws, size_t ws_size,
                              hipStream_t stream)
{
    (void)in_sizes; (void)n_in; (void)out_size; (void)ws_size;

    const float* x_d   = (const float*)d_in[0];
    const float* micF  = (const float*)d_in[1];
    const float* disF  = (const float*)d_in[2];
    const float* bn_b[3] = {(const float*)d_in[3], (const float*)d_in[5], (const float*)d_in[7]};
    const float* bn_g[3] = {(const float*)d_in[4], (const float*)d_in[6], (const float*)d_in[8]};
    const float* bnDis_b = (const float*)d_in[9];
    const float* bnDis_g = (const float*)d_in[10];
    const float* bnMic_b = (const float*)d_in[11];
    const float* bnMic_g = (const float*)d_in[12];
    const float* cWa[3] = {(const float*)d_in[13], (const float*)d_in[17], (const float*)d_in[21]};
    const float* cWb[3] = {(const float*)d_in[14], (const float*)d_in[18], (const float*)d_in[22]};
    const float* cba[3] = {(const float*)d_in[15], (const float*)d_in[19], (const float*)d_in[23]};
    const float* cbb[3] = {(const float*)d_in[16], (const float*)d_in[20], (const float*)d_in[24]};
    const float* disW  = (const float*)d_in[25];
    const float* disBi = (const float*)d_in[26];
    const float* fc1W  = (const float*)d_in[27];
    const float* fc1b  = (const float*)d_in[28];
    const float* micW  = (const float*)d_in[29];
    const float* micBi = (const float*)d_in[30];
    const int*   ei    = (const int*)d_in[31];
    const int*   batch = (const int*)d_in[32];
    const int*   src   = ei;
    const int*   dst   = ei + N_EDGES;
    float* out = (float*)d_out;

    // Workspace carve-up (all offsets 256B aligned).
    char* ws = (char*)d_ws;
    const size_t BIG = (size_t)N_NODES * ODIM * sizeof(float);        // 204.8 MB
    float*    bufA   = (float*)(ws + 0);
    float*    bufB   = (float*)(ws + BIG);
    char*     p      = ws + 2 * BIG;
    unsigned* pooled = (unsigned*)p;            p += (size_t)N_GRAPHS * ODIM * 4;
    float*    stats  = (float*)p;               p += 1024;
    _Float16* whA[3];
    _Float16* whB[3];
    for (int l = 0; l < 3; ++l) {
        whA[l] = (_Float16*)p;  p += (size_t)ODIM * 128 * 2;
        whB[l] = (_Float16*)p;  p += (size_t)ODIM * 128 * 2;
    }
    _Float16* whFc1 = (_Float16*)p;             p += (size_t)ODIM * 128 * 2;
    _Float16* whMic = (_Float16*)p;             p += (size_t)ODIM * 1024 * 2;
    _Float16* whDis = (_Float16*)p;             p += (size_t)ODIM * 1024 * 2;
    float*    preMic = (float*)p;               p += (size_t)4096 * ODIM * 4;
    float*    preDis = (float*)p;               /* += 4096*128*4 */

    const int nodeElemBlocks = (int)(((long long)N_NODES * ODIM + 255) / 256);  // 200000
    const int gemmBlocksNode = N_NODES / 128;                                   // 3125

    // --- pre-convert all weights to f16 transposed (WhT[n][k]) ---
    convert_wT_kernel<<<64,  256, 0, stream>>>(cWa[0], whA[0], 128, 78);
    convert_wT_kernel<<<64,  256, 0, stream>>>(cWb[0], whB[0], 128, 128);
    convert_wT_kernel<<<64,  256, 0, stream>>>(cWa[1], whA[1], 128, 128);
    convert_wT_kernel<<<64,  256, 0, stream>>>(cWb[1], whB[1], 128, 128);
    convert_wT_kernel<<<64,  256, 0, stream>>>(cWa[2], whA[2], 128, 128);
    convert_wT_kernel<<<64,  256, 0, stream>>>(cWb[2], whB[2], 128, 128);
    convert_wT_kernel<<<64,  256, 0, stream>>>(fc1W,   whFc1,  128, 128);
    convert_wT_kernel<<<512, 256, 0, stream>>>(micW,   whMic, 1024, 1024);
    convert_wT_kernel<<<512, 256, 0, stream>>>(disW,   whDis, 1024, 1024);

    // --- conv1 aggregation (78-wide, zero-padded to 128) ---
    init_agg78_kernel<<<nodeElemBlocks, 256, 0, stream>>>(x_d, bufA);
    scatter78_kernel<<<(int)(((long long)N_EDGES * 128 + 255) / 256), 256, 0, stream>>>(src, dst, x_d, bufA);

    const float* xin = nullptr;
    for (int l = 0; l < 3; ++l) {
        if (l > 0) {
            // agg = x + scatter(x): copy then atomic scatter-add.
            (void)hipMemcpyAsync(bufA, bufB, BIG, hipMemcpyDeviceToDevice, stream);
            scatter128_kernel<<<(int)(((long long)N_EDGES * 32 + 255) / 256), 256, 0, stream>>>(src, dst, bufB, bufA);
        }
        // h = relu(agg @ Wa + ba)
        gemm_wmma_kernel<<<gemmBlocksNode, 256, 0, stream>>>(bufA, ODIM, whA[l], cba[l], bufB, ODIM,
                                                             N_NODES, ODIM, 1);
        // pre = h @ Wb + bb
        gemm_wmma_kernel<<<gemmBlocksNode, 256, 0, stream>>>(bufB, ODIM, whB[l], cbb[l], bufA, ODIM,
                                                             N_NODES, ODIM, 0);
        // x = BN(relu(pre))
        zero_stats_kernel<<<1, 256, 0, stream>>>(stats);
        bn_stats_relu_kernel<<<(N_NODES + 1023) / 1024, 128, 0, stream>>>(bufA, stats, N_NODES, 1024);
        bn_apply_relu_kernel<<<nodeElemBlocks, 256, 0, stream>>>(bufA, stats, bn_g[l], bn_b[l], bufB, N_NODES);
        xin = bufB;
    }

    // --- segment max pool + fc1 ---
    pool_init_kernel<<<(N_GRAPHS * ODIM + 255) / 256, 256, 0, stream>>>(pooled);
    pool_max_kernel<<<nodeElemBlocks, 256, 0, stream>>>(xin, batch, pooled);
    pool_decode_kernel<<<(N_GRAPHS * ODIM + 255) / 256, 256, 0, stream>>>(pooled);
    gemm_wmma_kernel<<<N_GRAPHS / 128, 256, 0, stream>>>((const float*)pooled, ODIM, whFc1, fc1b,
                                                         out, ODIM, N_GRAPHS, ODIM, 1);

    // --- mic branch: BN(relu(mic @ W + b)) ---
    gemm_wmma_kernel<<<4096 / 128, 256, 0, stream>>>(micF, 1024, whMic, micBi, preMic, ODIM,
                                                     4096, 1024, 0);
    zero_stats_kernel<<<1, 256, 0, stream>>>(stats);
    bn_stats_relu_kernel<<<(4096 + 1023) / 1024, 128, 0, stream>>>(preMic, stats, 4096, 1024);
    bn_apply_relu_kernel<<<(4096 * ODIM + 255) / 256, 256, 0, stream>>>(
        preMic, stats, bnMic_g, bnMic_b, out + (size_t)N_GRAPHS * ODIM, 4096);

    // --- dis branch ---
    gemm_wmma_kernel<<<4096 / 128, 256, 0, stream>>>(disF, 1024, whDis, disBi, preDis, ODIM,
                                                     4096, 1024, 0);
    zero_stats_kernel<<<1, 256, 0, stream>>>(stats);
    bn_stats_relu_kernel<<<(4096 + 1023) / 1024, 128, 0, stream>>>(preDis, stats, 4096, 1024);
    bn_apply_relu_kernel<<<(4096 * ODIM + 255) / 256, 256, 0, stream>>>(
        preDis, stats, bnDis_g, bnDis_b, out + (size_t)N_GRAPHS * ODIM + (size_t)4096 * ODIM, 4096);
}
